// Attention_12197707121071
// MI455X (gfx1250) — matrix operand order repaired and checked by reading, not compiled, not run on hardware
//
#include <hip/hip_runtime.h>
#include <hip/hip_bf16.h>

// ---------------------------------------------------------------------------
// GQA attention block for MI455X (gfx1250, wave32, WMMA).
// Matmuls via v_wmma_f32_16x16x32_bf16 (f32 accumulate); pure-copy tile
// staging via global_load_async_to_lds_b128 (ASYNCcnt) where no conversion
// is needed.
// ---------------------------------------------------------------------------

typedef __attribute__((ext_vector_type(16))) __bf16 v16bf;
typedef __attribute__((ext_vector_type(8)))  float  v8f;
typedef unsigned short u16;

union AFrag { v16bf v; uint4 q[2]; };   // 16 bf16 = 32B = 8 VGPRs

static __device__ __forceinline__ u16 f2bf(float f) {
    union { __hip_bfloat16 h; u16 u; } x; x.h = __float2bfloat16(f); return x.u;
}
static __device__ __forceinline__ float bf2f(u16 u) {
    union { __hip_bfloat16 h; u16 u; } x; x.u = u; return __bfloat162float(x.h);
}

// Async 16-byte global->LDS copy (per-lane addresses).  Generic pointers to
// __shared__ carry the wave-relative LDS offset in their low 32 bits
// (ISA 10.2: LDS aperture).
static __device__ __forceinline__ void async_copy_b128(void* lds, const void* gaddr) {
    unsigned l = (unsigned)(size_t)lds;
    asm volatile("global_load_async_to_lds_b128 %0, %1, off"
                 :: "v"(l), "v"(gaddr) : "memory");
}
static __device__ __forceinline__ void async_wait0() {
    asm volatile("s_wait_asynccnt 0" ::: "memory");
}

#define DIM       2048
#define SEQ       2048
#define NH        16
#define NKV       4
#define HD        128
#define LDQKV     3072      // q(2048) | k(512) | v(512) fused, bf16
#define NEG_BIG   (-3.0e38f)
#define SM_SCALE  0.08838834764831845f  // 1/sqrt(128)

// ---------------------------------------------------------------------------
// Tiled GEMM:  C[M,N] = A[M,K] * B[K,N]
//   A: f32 or bf16 (template), B: f32 (weights), C: bf16 or f32 (template)
//   Workgroup tile 128x128, 8 waves (4x2), each wave 32x64 = 2x4 WMMA accums.
//   Double-buffered LDS (2 x 16KB):
//     bf16 A tiles:  async global->LDS copy straight into the next buffer
//     f32 tiles:     register-staged (conversion in flight)
//   One barrier per K iteration; s_wait_asynccnt 0 before the barrier.
// ---------------------------------------------------------------------------
template<bool A_BF16, bool OUT_BF16>
__global__ __launch_bounds__(256)
void gemm_wmma(const void* __restrict__ Ap, const float* __restrict__ Bp,
               void* __restrict__ Cp, int K, int lda, int ldb, int ldc)
{
    __shared__ __align__(16) u16 As[2][128 * 32];   // [m][k]
    __shared__ __align__(16) u16 Bs[2][128 * 32];   // [n][k]

    const int tid   = threadIdx.x;
    const int wave  = tid >> 5;
    const int lane  = tid & 31;
    const int lhalf = lane & 15;
    const int khi   = lane >> 4;          // 0: lanes 0-15, 1: lanes 16-31
    const int wm    = wave >> 1;          // 0..3  (rows, 32 each)
    const int wn    = wave & 1;           // 0..1  (cols, 64 each)
    const int m0    = blockIdx.x * 128;
    const int n0    = blockIdx.y * 128;
    const int NT    = K >> 5;             // number of 32-wide K tiles

    v8f acc[2][4];
#pragma unroll
    for (int i = 0; i < 2; ++i)
#pragma unroll
        for (int j = 0; j < 4; ++j)
            acc[i][j] = (v8f){0.f,0.f,0.f,0.f,0.f,0.f,0.f,0.f};

    // ---- prologue: stage tile 0 into LDS buffer 0 ------------------------
    {
        const int k0 = 0;
        if (A_BF16) {
            const u16* A = (const u16*)Ap;
#pragma unroll
            for (int j = 0; j < 2; ++j) {
                int idx = tid + j * 256, m = idx >> 2, c = idx & 3;
                async_copy_b128(As[0] + m * 32 + c * 8,
                                A + (size_t)(m0 + m) * lda + k0 + c * 8);
            }
        } else {
            const float* A = (const float*)Ap;
#pragma unroll
            for (int j = 0; j < 4; ++j) {
                int idx = tid + j * 256, m = idx >> 3, c = idx & 7;
                float4 v = *(const float4*)(A + (size_t)(m0 + m) * lda + k0 + c * 4);
                uint2 w;
                w.x = (unsigned)f2bf(v.x) | ((unsigned)f2bf(v.y) << 16);
                w.y = (unsigned)f2bf(v.z) | ((unsigned)f2bf(v.w) << 16);
                *(uint2*)(As[0] + m * 32 + c * 4) = w;
            }
        }
#pragma unroll
        for (int j = 0; j < 4; ++j) {
            int idx = tid + j * 256, kk = idx >> 5, nc = idx & 31;
            float4 v = *(const float4*)(Bp + (size_t)(k0 + kk) * ldb + n0 + nc * 4);
            int nb = nc * 4;
            Bs[0][(nb + 0) * 32 + kk] = f2bf(v.x);
            Bs[0][(nb + 1) * 32 + kk] = f2bf(v.y);
            Bs[0][(nb + 2) * 32 + kk] = f2bf(v.z);
            Bs[0][(nb + 3) * 32 + kk] = f2bf(v.w);
        }
        if (A_BF16) async_wait0();
    }
    __syncthreads();

    for (int kt = 0; kt < NT; ++kt) {
        const int cur = kt & 1, nxt = cur ^ 1;
        const bool has_next = (kt + 1) < NT;

        // ---- issue next tile's loads (async -> LDS, or into registers) ---
        float4 aR[4]; float4 bR[4];
        if (has_next) {
            const int k0 = (kt + 1) << 5;
            if (A_BF16) {
                const u16* A = (const u16*)Ap;
#pragma unroll
                for (int j = 0; j < 2; ++j) {
                    int idx = tid + j * 256, m = idx >> 2, c = idx & 3;
                    async_copy_b128(As[nxt] + m * 32 + c * 8,
                                    A + (size_t)(m0 + m) * lda + k0 + c * 8);
                }
            } else {
                const float* A = (const float*)Ap;
#pragma unroll
                for (int j = 0; j < 4; ++j) {
                    int idx = tid + j * 256, m = idx >> 3, c = idx & 7;
                    aR[j] = *(const float4*)(A + (size_t)(m0 + m) * lda + k0 + c * 4);
                }
            }
#pragma unroll
            for (int j = 0; j < 4; ++j) {
                int idx = tid + j * 256, kk = idx >> 5, nc = idx & 31;
                bR[j] = *(const float4*)(Bp + (size_t)(k0 + kk) * ldb + n0 + nc * 4);
            }
            // prefetch tile kt+2 toward L2/L0 while WMMAs run
            if (kt + 2 < NT) {
                const int k2 = (kt + 2) << 5;
                if (A_BF16) {
                    const u16* A = (const u16*)Ap;
                    __builtin_prefetch(A + (size_t)(m0 + (tid >> 2)) * lda + k2, 0, 3);
                } else {
                    const float* A = (const float*)Ap;
                    __builtin_prefetch(A + (size_t)(m0 + (tid >> 3)) * lda + k2, 0, 3);
                }
                __builtin_prefetch(Bp + (size_t)(k2 + (tid >> 5)) * ldb + n0 +
                                   (tid & 31) * 4, 0, 3);
            }
        }

        // ---- compute current tile (ISA 16-bit A/B fragment layouts) ------
        AFrag a[2];
#pragma unroll
        for (int mb = 0; mb < 2; ++mb) {
            int row = wm * 32 + mb * 16 + lhalf;
            int kb  = khi * 8;                       // A: K = kb..kb+7, kb+16..kb+23
            a[mb].q[0] = *(const uint4*)(As[cur] + row * 32 + kb);
            a[mb].q[1] = *(const uint4*)(As[cur] + row * 32 + kb + 16);
        }
        AFrag bfr[4];
#pragma unroll
        for (int nb = 0; nb < 4; ++nb) {
            int col = wn * 64 + nb * 16 + lhalf;
            int kb  = khi * 8;                       // B: K = kb..kb+7, kb+16..kb+23
            bfr[nb].q[0] = *(const uint4*)(Bs[cur] + col * 32 + kb);
            bfr[nb].q[1] = *(const uint4*)(Bs[cur] + col * 32 + kb + 16);
        }
#pragma unroll
        for (int mb = 0; mb < 2; ++mb)
#pragma unroll
            for (int nb = 0; nb < 4; ++nb)
                acc[mb][nb] = __builtin_amdgcn_wmma_f32_16x16x32_bf16(
                    false, a[mb].v, false, bfr[nb].v, (short)0, acc[mb][nb],
                    false, false);

        // ---- convert + write next tile to alternate LDS buffer -----------
        if (has_next) {
            if (!A_BF16) {
#pragma unroll
                for (int j = 0; j < 4; ++j) {
                    int idx = tid + j * 256, m = idx >> 3, c = idx & 7;
                    uint2 w;
                    w.x = (unsigned)f2bf(aR[j].x) | ((unsigned)f2bf(aR[j].y) << 16);
                    w.y = (unsigned)f2bf(aR[j].z) | ((unsigned)f2bf(aR[j].w) << 16);
                    *(uint2*)(As[nxt] + m * 32 + c * 4) = w;
                }
            }
#pragma unroll
            for (int j = 0; j < 4; ++j) {
                int idx = tid + j * 256, kk = idx >> 5, nc = idx & 31;
                int nb = nc * 4;
                Bs[nxt][(nb + 0) * 32 + kk] = f2bf(bR[j].x);
                Bs[nxt][(nb + 1) * 32 + kk] = f2bf(bR[j].y);
                Bs[nxt][(nb + 2) * 32 + kk] = f2bf(bR[j].z);
                Bs[nxt][(nb + 3) * 32 + kk] = f2bf(bR[j].w);
            }
        }
        if (A_BF16) async_wait0();   // drain our async LDS writes pre-barrier
        __syncthreads();
    }

    // ---- epilogue: C layout -> global -----------------------------------
#pragma unroll
    for (int mb = 0; mb < 2; ++mb)
#pragma unroll
        for (int nb = 0; nb < 4; ++nb)
#pragma unroll
            for (int i = 0; i < 8; ++i) {
                int row = m0 + wm * 32 + mb * 16 + i + khi * 8;
                int col = n0 + wn * 64 + nb * 16 + lhalf;
                if (OUT_BF16)
                    ((u16*)Cp)[(size_t)row * ldc + col] = f2bf(acc[mb][nb][i]);
                else
                    ((float*)Cp)[(size_t)row * ldc + col] = acc[mb][nb][i];
            }
}

// ---------------------------------------------------------------------------
// RoPE applied in place to the q/k columns of the fused bf16 qkv buffer.
// One thread per (row, head, freq-pair). 20 heads = 16 q + 4 k.
// ---------------------------------------------------------------------------
__global__ __launch_bounds__(256)
void rope_kernel(u16* __restrict__ qkv, const float* __restrict__ fc,
                 const float* __restrict__ fs)
{
    int idx = blockIdx.x * blockDim.x + threadIdx.x;
    const int total = 4096 * 20 * 64;
    if (idx >= total) return;
    int p    = idx & 63;
    int t    = idx >> 6;
    int head = t % 20;
    int r    = t / 20;
    int s    = r & (SEQ - 1);
    int col  = (head < 16) ? head * HD + 2 * p
                           : DIM + (head - 16) * HD + 2 * p;
    size_t off = (size_t)r * LDQKV + col;
    float xr = bf2f(qkv[off]), xi = bf2f(qkv[off + 1]);
    float c = fc[s * 64 + p], sn = fs[s * 64 + p];
    qkv[off]     = f2bf(xr * c - xi * sn);
    qkv[off + 1] = f2bf(xr * sn + xi * c);
}

// ---------------------------------------------------------------------------
// Flash attention, causal, GQA.  Grid = (S/64, NH, B), block = 128 (4 waves).
// Wave w owns query rows [64*qt + 16w, +16).  Key tiles of 64, online softmax.
// Q and K tiles staged with global_load_async_to_lds_b128; V transposed via
// b128 global loads + element LDS stores.  Next K/V tile prefetched.
// Scores and PV both via v_wmma bf16; P goes C-layout -> A-layout through a
// per-wave LDS staging buffer.
// ---------------------------------------------------------------------------
__global__ __launch_bounds__(128)
void attn_kernel(const u16* __restrict__ qkv, u16* __restrict__ ab)
{
    __shared__ __align__(16) u16 Qs[64 * 128];
    __shared__ __align__(16) u16 Ks[64 * 128];     // [t][d]  (B for Q*K^T)
    __shared__ __align__(16) u16 Vt[128 * 64];     // [d][t]  (B for P*V)
    __shared__ __align__(16) u16 Ps[4 * 16 * 80];  // per-wave P staging

    const int tid   = threadIdx.x;
    const int wave  = tid >> 5;
    const int lane  = tid & 31;
    const int lhalf = lane & 15;
    const int khi   = lane >> 4;
    const int qt    = blockIdx.x;
    const int h     = blockIdx.y;
    const int b     = blockIdx.z;
    const int g     = h >> 2;                       // kv head
    const int qbase = qt * 64;
    const size_t rowbase = (size_t)b * SEQ;

    // ---- async-load Q tile (64 x 128 bf16) ------------------------------
#pragma unroll
    for (int idx = tid; idx < 64 * 16; idx += 128) {
        int r = idx >> 4, c = idx & 15;
        async_copy_b128(Qs + r * 128 + c * 8,
                        qkv + (rowbase + qbase + r) * LDQKV + h * HD + c * 8);
    }
    async_wait0();
    __syncthreads();

    AFrag qf[4];
#pragma unroll
    for (int c = 0; c < 4; ++c) {
        int row = wave * 16 + lhalf;
        int kb  = c * 32 + khi * 8;
        qf[c].q[0] = *(const uint4*)(Qs + row * 128 + kb);
        qf[c].q[1] = *(const uint4*)(Qs + row * 128 + kb + 16);
    }

    v8f o[8];
#pragma unroll
    for (int i = 0; i < 8; ++i) o[i] = (v8f){0.f,0.f,0.f,0.f,0.f,0.f,0.f,0.f};
    float mrow[8], lrow[8];
#pragma unroll
    for (int i = 0; i < 8; ++i) { mrow[i] = NEG_BIG; lrow[i] = 0.f; }

    for (int tb = 0; tb <= qt; ++tb) {
        const int tbase = tb * 64;
        __syncthreads();
        // ---- K tile: async global->LDS straight copy --------------------
#pragma unroll
        for (int idx = tid; idx < 64 * 16; idx += 128) {
            int t = idx >> 4, c = idx & 15;
            async_copy_b128(Ks + t * 128 + c * 8,
                            qkv + (rowbase + tbase + t) * LDQKV +
                            DIM + g * HD + c * 8);
        }
        // ---- V tile: b128 global loads, transposed element stores -------
#pragma unroll
        for (int idx = tid; idx < 64 * 16; idx += 128) {
            int t = idx >> 4, c = idx & 15;
            union { uint4 q; u16 e[8]; } u;
            u.q = *(const uint4*)(qkv + (rowbase + tbase + t) * LDQKV +
                                  DIM + NKV * HD + g * HD + c * 8);
#pragma unroll
            for (int j = 0; j < 8; ++j)
                Vt[(c * 8 + j) * 64 + t] = u.e[j];
        }
        // ---- prefetch next K/V tile (global_prefetch_b8) ----------------
        if (tb < qt) {
            const int tn = tbase + 64 + (tid >> 1);
            const int half = (tid & 1) * 64;
            __builtin_prefetch(qkv + (rowbase + tn) * LDQKV + DIM + g * HD + half,
                               0, 3);
            __builtin_prefetch(qkv + (rowbase + tn) * LDQKV + DIM + NKV * HD +
                               g * HD + half, 0, 3);
        }
        async_wait0();   // K-tile async writes visible before barrier
        __syncthreads();

        // ---- S = Q K^T  (16 WMMAs) --------------------------------------
        v8f s[4];
#pragma unroll
        for (int nb = 0; nb < 4; ++nb)
            s[nb] = (v8f){0.f,0.f,0.f,0.f,0.f,0.f,0.f,0.f};
#pragma unroll
        for (int nb = 0; nb < 4; ++nb)
#pragma unroll
            for (int c = 0; c < 4; ++c) {
                AFrag kf;
                int col = nb * 16 + lhalf;
                int kb  = c * 32 + khi * 8;
                kf.q[0] = *(const uint4*)(Ks + col * 128 + kb);
                kf.q[1] = *(const uint4*)(Ks + col * 128 + kb + 16);
                s[nb] = __builtin_amdgcn_wmma_f32_16x16x32_bf16(
                    false, qf[c].v, false, kf.v, (short)0, s[nb], false, false);
            }

        // ---- scale + causal mask + row max ------------------------------
        const bool diag = (tb == qt);
        float mt[8];
#pragma unroll
        for (int i = 0; i < 8; ++i) mt[i] = NEG_BIG;
#pragma unroll
        for (int nb = 0; nb < 4; ++nb)
#pragma unroll
            for (int i = 0; i < 8; ++i) {
                float v = s[nb][i] * SM_SCALE;
                if (diag) {
                    int rloc = wave * 16 + i + khi * 8;
                    int tloc = nb * 16 + lhalf;
                    if (tloc > rloc) v = NEG_BIG;
                }
                s[nb][i] = v;
                mt[i] = fmaxf(mt[i], v);
            }
#pragma unroll
        for (int i = 0; i < 8; ++i) {
#pragma unroll
            for (int off = 1; off < 16; off <<= 1)
                mt[i] = fmaxf(mt[i], __shfl_xor(mt[i], off, 16));
        }

        // ---- online softmax update --------------------------------------
        float alpha[8];
#pragma unroll
        for (int i = 0; i < 8; ++i) {
            float mn = fmaxf(mrow[i], mt[i]);
            alpha[i] = __expf(mrow[i] - mn);
            mrow[i]  = mn;
        }
#pragma unroll
        for (int nb = 0; nb < 8; ++nb)
#pragma unroll
            for (int i = 0; i < 8; ++i) o[nb][i] *= alpha[i];

        float psum[8];
#pragma unroll
        for (int i = 0; i < 8; ++i) psum[i] = 0.f;
#pragma unroll
        for (int nb = 0; nb < 4; ++nb)
#pragma unroll
            for (int i = 0; i < 8; ++i) {
                float p = __expf(s[nb][i] - mrow[i]);
                psum[i] += p;
                Ps[wave * 1280 + (i + khi * 8) * 80 + nb * 16 + lhalf] = f2bf(p);
            }
#pragma unroll
        for (int i = 0; i < 8; ++i) lrow[i] = lrow[i] * alpha[i] + psum[i];
        __syncthreads();   // P: C-layout writes -> A-layout reads

        // ---- O += P V  (16 WMMAs) ---------------------------------------
        AFrag pf[2];
#pragma unroll
        for (int c = 0; c < 2; ++c) {
            int base = wave * 1280 + lhalf * 80 + c * 32 + khi * 8;
            pf[c].q[0] = *(const uint4*)(Ps + base);
            pf[c].q[1] = *(const uint4*)(Ps + base + 16);
        }
#pragma unroll
        for (int nb = 0; nb < 8; ++nb)
#pragma unroll
            for (int c = 0; c < 2; ++c) {
                AFrag vf;
                int col = nb * 16 + lhalf;
                int kb  = c * 32 + khi * 8;
                vf.q[0] = *(const uint4*)(Vt + col * 64 + kb);
                vf.q[1] = *(const uint4*)(Vt + col * 64 + kb + 16);
                o[nb] = __builtin_amdgcn_wmma_f32_16x16x32_bf16(
                    false, pf[c].v, false, vf.v, (short)0, o[nb], false, false);
            }
    }

    // ---- finalize: reduce l across the 16-lane row group, write bf16 ----
#pragma unroll
    for (int i = 0; i < 8; ++i) {
#pragma unroll
        for (int off = 1; off < 16; off <<= 1)
            lrow[i] += __shfl_xor(lrow[i], off, 16);
    }
#pragma unroll
    for (int nb = 0; nb < 8; ++nb)
#pragma unroll
        for (int i = 0; i < 8; ++i) {
            int r   = qbase + wave * 16 + i + khi * 8;
            int col = h * HD + nb * 16 + lhalf;
            ab[(rowbase + r) * (size_t)DIM + col] = f2bf(o[nb][i] / lrow[i]);
        }
}

// ---------------------------------------------------------------------------
// Host-side orchestration.
//   d_ws layout:  [0, 24 MB)  fused qkv bf16 (4096 x 3072)
//                 [24 MB, +16 MB)  attention output bf16 (4096 x 2048)
// ---------------------------------------------------------------------------
extern "C" void kernel_launch(void* const* d_in, const int* in_sizes, int n_in,
                              void* d_out, int out_size, void* d_ws, size_t ws_size,
                              hipStream_t stream)
{
    const float* x  = (const float*)d_in[0];
    const float* fc = (const float*)d_in[1];
    const float* fs = (const float*)d_in[2];
    const float* wq = (const float*)d_in[3];
    const float* wk = (const float*)d_in[4];
    const float* wv = (const float*)d_in[5];
    const float* wo = (const float*)d_in[6];

    u16* qkv = (u16*)d_ws;
    u16* ab  = (u16*)((char*)d_ws + (size_t)4096 * LDQKV * sizeof(u16));

    // QKV projections -> fused bf16 buffer (ldc = 3072)
    gemm_wmma<false, true><<<dim3(32, 16), 256, 0, stream>>>(
        x, wq, qkv + 0, DIM, DIM, DIM, LDQKV);
    gemm_wmma<false, true><<<dim3(32, 4), 256, 0, stream>>>(
        x, wk, qkv + DIM, DIM, DIM, NKV * HD, LDQKV);
    gemm_wmma<false, true><<<dim3(32, 4), 256, 0, stream>>>(
        x, wv, qkv + DIM + NKV * HD, DIM, DIM, NKV * HD, LDQKV);

    // RoPE on q and k columns in place
    const int rope_total = 4096 * 20 * 64;
    rope_kernel<<<(rope_total + 255) / 256, 256, 0, stream>>>(qkv, fc, fs);

    // Flash attention
    attn_kernel<<<dim3(SEQ / 64, NH, 2), 128, 0, stream>>>(qkv, ab);

    // Output projection -> f32 d_out
    gemm_wmma<true, false><<<dim3(32, 16), 256, 0, stream>>>(
        ab, wo, d_out, DIM, DIM, DIM, DIM);
}
